// AttentionHead_695784702263
// MI455X (gfx1250) — matrix operand order repaired and checked by reading, not compiled, not run on hardware
//
#include <hip/hip_runtime.h>
#include <hip/hip_bf16.h>

typedef __bf16 bf16;
typedef __attribute__((ext_vector_type(16))) __bf16 v16bf;
typedef __attribute__((ext_vector_type(8)))  __bf16 v8bf;
typedef __attribute__((ext_vector_type(8)))  float  v8f;
typedef int v4i __attribute__((vector_size(16)));

static constexpr int kB = 4;
static constexpr int kT = 4096;
static constexpr int kD = 256;
static constexpr float kScale = 0.0625f; // 1/sqrt(256)

#define WMMA_BF16(a, b, c) \
  __builtin_amdgcn_wmma_f32_16x16x32_bf16(false, (a), false, (b), (short)0, (c), false, false)

#ifndef __has_builtin
#define __has_builtin(x) 0
#endif

#if __has_builtin(__builtin_amdgcn_global_load_async_to_lds_b128)
#define HAVE_ASYNC_LDS 1
#else
#define HAVE_ASYNC_LDS 0
#endif

// 16-byte global -> LDS copy: async (ASYNCcnt-tracked) when available.
// Prototype (from hipcc diagnostic): param0 = v4i32 addrspace(1)*,
// param1 = v4i32 addrspace(3)*, then imm offset, imm cpol.
__device__ __forceinline__ void copy16_g2l(const bf16* gsrc, bf16* ldst) {
#if HAVE_ASYNC_LDS
  __builtin_amdgcn_global_load_async_to_lds_b128(
      (__attribute__((address_space(1))) v4i*)gsrc,
      (__attribute__((address_space(3))) v4i*)ldst, 0, 0);
#else
  *(v8bf*)ldst = *(const v8bf*)gsrc;
#endif
}

__device__ __forceinline__ void wait_async_done() {
#if __has_builtin(__builtin_amdgcn_s_wait_asynccnt)
  __builtin_amdgcn_s_wait_asynccnt(0);
#else
  asm volatile("s_wait_asynccnt 0x0" ::: "memory");
#endif
}

// ---------------------------------------------------------------------------
// Kernel 1: qkv = x @ W^T + bias, de-interleaved (col j -> n=j%3, d=j/3),
// written as bf16 Q/K/V [B*T, D]. One wave computes a 16x16 output tile,
// accumulating over K=256 in 8 WMMA steps (f32 -> bf16 convert in-register).
// ---------------------------------------------------------------------------
__global__ __launch_bounds__(32) void qkv_proj_kernel(
    const float* __restrict__ x,    // [B*T, D]
    const float* __restrict__ W,    // [3D, D]
    const float* __restrict__ bias, // [3D]
    bf16* __restrict__ Qw, bf16* __restrict__ Kw, bf16* __restrict__ Vw) {
  const int lane = threadIdx.x & 31;
  const int half = lane >> 4;  // 0/1: which 16-lane group
  const int l16  = lane & 15;
  const int m0 = blockIdx.x * 16;  // row tile base in [0, B*T)
  const int n0 = blockIdx.y * 16;  // col tile base in [0, 768)

  const float* xrow = x + (size_t)(m0 + l16) * kD;      // A: row m0+l16
  const float* wrow = W + (size_t)(n0 + l16) * kD;      // B: col j = n0+l16 -> W row j

  v8f acc = {};
#pragma unroll
  for (int c = 0; c < 8; ++c) {
    // A fragment (16x32 bf16): halves 0..7 -> k = c*32 + half*8 + i; 8..15 -> +16
    v16bf a, b;
    const int ka = c * 32 + half * 8;
#pragma unroll
    for (int i = 0; i < 8; ++i) {
      a[i]     = (bf16)xrow[ka + i];
      a[i + 8] = (bf16)xrow[ka + 16 + i];
    }
    // B fragment (32x16 bf16): halves 0..7 -> k = c*32 + half*8 + i; 8..15 -> +16
    const int kb = c * 32 + half * 8;
#pragma unroll
    for (int i = 0; i < 16; ++i) b[i] = (bf16)wrow[kb + i + (i & 8)];
    acc = WMMA_BF16(a, b, acc);
  }

  const int j = n0 + l16;        // this lane's output column
  const float bj = bias[j];
  const int nsel = j % 3;        // 0->Q, 1->K, 2->V (stride-3 interleave)
  const int dcol = j / 3;
  bf16* dst = (nsel == 0) ? Qw : (nsel == 1) ? Kw : Vw;
#pragma unroll
  for (int r = 0; r < 8; ++r) {
    const int row = m0 + r + 8 * half;  // C/D layout: lane group selects M half
    dst[(size_t)row * kD + dcol] = (bf16)(acc[r] + bj);
  }
}

// ---------------------------------------------------------------------------
// Kernel 2: flash attention. 4 waves/block; wave owns 16 query rows with the
// full D=256 accumulator in registers (16 x v8f). Double-buffered 32-key
// tiles: K staged with async global->LDS copies (ASYNCcnt), V staged
// transposed so P*V B-fragments are contiguous LDS reads.
// ---------------------------------------------------------------------------
static constexpr int kTQ = 16;        // q rows per wave
static constexpr int kNW = 4;         // waves per block
static constexpr int kTJ = 32;        // keys per tile
static constexpr int kKP = kD + 16;   // Klds row stride (halves), bank spread
static constexpr int kVP = kTJ + 8;   // Vlds row stride
static constexpr int kPP = kTJ + 8;   // Plds row stride

__global__ __launch_bounds__(128) void attn_kernel(
    const bf16* __restrict__ Q, const bf16* __restrict__ K,
    const bf16* __restrict__ V, const int* __restrict__ mask,
    float* __restrict__ out) {
  __shared__ __align__(16) bf16 Klds[2][kTJ][kKP];      // [buf][key][d]
  __shared__ __align__(16) bf16 Vlds[2][kD][kVP];       // [buf][d][key] (transposed)
  __shared__ __align__(16) bf16 Plds[kNW][kTQ][kPP];    // per-wave P bounce

  const int tid  = threadIdx.x;
  const int lane = tid & 31;
  const int w    = tid >> 5;
  const int half = lane >> 4;
  const int l16  = lane & 15;

  const int qblocks = kT / (kTQ * kNW);                 // 64 per batch
  const int b  = blockIdx.x / qblocks;
  const int q0 = (blockIdx.x % qblocks) * (kTQ * kNW) + w * kTQ;

  const bf16* Qb = Q + (size_t)b * kT * kD;
  const bf16* Kb = K + (size_t)b * kT * kD;
  const bf16* Vb = V + (size_t)b * kT * kD;
  const int*  mb = mask + b * kT;

  // Q strip (16 x 256) as 8 A-fragments, resident in VGPRs for all 128 tiles.
  v16bf qf[8];
  {
    const bf16* qrow = Qb + (size_t)(q0 + l16) * kD;
    __builtin_prefetch(qrow, 0, 3);   // global_prefetch_b8
#pragma unroll
    for (int c = 0; c < 8; ++c) {
      const v8bf lo = *(const v8bf*)(qrow + c * 32 + half * 16);
      const v8bf hi = *(const v8bf*)(qrow + c * 32 + half * 16 + 8);
#pragma unroll
      for (int i = 0; i < 8; ++i) { qf[c][i] = lo[i]; qf[c][i + 8] = hi[i]; }
    }
  }

  v8f O[16];
#pragma unroll
  for (int dt = 0; dt < 16; ++dt)
#pragma unroll
    for (int i = 0; i < 8; ++i) O[dt][i] = 0.0f;

  float m_run[8], l_run[8];
#pragma unroll
  for (int r = 0; r < 8; ++r) { m_run[r] = -1e30f; l_run[r] = 0.0f; }

  // ---- staging helpers (inlined lambdas) ----
  auto stageK = [&](int jk, int bi) {
    // 32 keys x 512 B = 1024 x 16 B chunks; 8 per thread, async when available.
#pragma unroll
    for (int it = 0; it < 8; ++it) {
      const int s = tid + it * 128;
      const int key = s >> 5, c16 = (s & 31) * 8;  // halves offset within row
      copy16_g2l(Kb + (size_t)(jk + key) * kD + c16, &Klds[bi][key][c16]);
    }
  };
  auto stageV = [&](int jk, int bi) {
    // transpose scatter: [key][d] -> [d][key]
#pragma unroll
    for (int it = 0; it < 8; ++it) {
      const int s = tid + it * 128;
      const int key = s >> 5, d0 = (s & 31) * 8;
      const v8bf vv = *(const v8bf*)(Vb + (size_t)(jk + key) * kD + d0);
#pragma unroll
      for (int i = 0; i < 8; ++i) Vlds[bi][d0 + i][key] = vv[i];
    }
  };

  // prologue: fill buffer 0
  stageK(0, 0);
  stageV(0, 0);
  wait_async_done();
  __syncthreads();

  const int ntiles = kT / kTJ;  // 128
  for (int jt = 0; jt < ntiles; ++jt) {
    const int cur = jt & 1, nxt = cur ^ 1;
    const int j0 = jt * kTJ;

    // prefetch next tile into the other buffer while we compute this one
    if (jt + 1 < ntiles) {
      stageK(j0 + kTJ, nxt);
      stageV(j0 + kTJ, nxt);
    }

    // ---- S = Q K^T for 32 keys: two 16x16 f32 accumulators ----
    v8f S0 = {}, S1 = {};
#pragma unroll
    for (int c = 0; c < 8; ++c) {
      v16bf b0, b1;
      const bf16* p0 = &Klds[cur][l16][c * 32 + half * 16];       // keys 0..15
      const bf16* p1 = &Klds[cur][16 + l16][c * 32 + half * 16];  // keys 16..31
      const v8bf a0 = *(const v8bf*)p0, a1 = *(const v8bf*)(p0 + 8);
      const v8bf c0 = *(const v8bf*)p1, c1 = *(const v8bf*)(p1 + 8);
#pragma unroll
      for (int i = 0; i < 8; ++i) {
        b0[i] = a0[i]; b0[i + 8] = a1[i];
        b1[i] = c0[i]; b1[i + 8] = c1[i];
      }
      S0 = WMMA_BF16(qf[c], b0, S0);
      S1 = WMMA_BF16(qf[c], b1, S1);
    }

    // additive mask bias for this lane's two key columns
    const float add0 = mb[j0 + l16]      ? 0.0f : -1e30f;
    const float add1 = mb[j0 + 16 + l16] ? 0.0f : -1e30f;

    // ---- online softmax (row = r + 8*half; row striped over 16 lanes) ----
    float e0[8], e1[8], alpha[8];
#pragma unroll
    for (int r = 0; r < 8; ++r) {
      const float s0 = S0[r] * kScale + add0;
      const float s1 = S1[r] * kScale + add1;
      float mx = fmaxf(s0, s1);
      mx = fmaxf(mx, __shfl_xor(mx, 1));
      mx = fmaxf(mx, __shfl_xor(mx, 2));
      mx = fmaxf(mx, __shfl_xor(mx, 4));
      mx = fmaxf(mx, __shfl_xor(mx, 8));
      const float mnew = fmaxf(m_run[r], mx);
      const float al = __expf(m_run[r] - mnew);
      m_run[r] = mnew;
      const float x0 = __expf(s0 - mnew);
      const float x1 = __expf(s1 - mnew);
      float rs = x0 + x1;
      rs += __shfl_xor(rs, 1);
      rs += __shfl_xor(rs, 2);
      rs += __shfl_xor(rs, 4);
      rs += __shfl_xor(rs, 8);
      l_run[r] = l_run[r] * al + rs;
      alpha[r] = al;
      e0[r] = x0; e1[r] = x1;
    }

    // rescale running O
#pragma unroll
    for (int dt = 0; dt < 16; ++dt)
#pragma unroll
      for (int r = 0; r < 8; ++r) O[dt][r] *= alpha[r];

    // ---- bounce P through LDS to re-layout as WMMA A fragment ----
#pragma unroll
    for (int r = 0; r < 8; ++r) {
      const int row = r + 8 * half;
      Plds[w][row][l16]      = (bf16)e0[r];
      Plds[w][row][16 + l16] = (bf16)e1[r];
    }
    asm volatile("s_wait_dscnt 0" ::: "memory");  // own-wave LDS RAW

    v16bf pf;
    {
      const bf16* pr = &Plds[w][l16][half * 16];
      const v8bf lo = *(const v8bf*)pr;
      const v8bf hi = *(const v8bf*)(pr + 8);
#pragma unroll
      for (int i = 0; i < 8; ++i) { pf[i] = lo[i]; pf[i + 8] = hi[i]; }
    }

    // ---- O += P (16x32) * V (32x256) over 16 d-tiles ----
#pragma unroll
    for (int dt = 0; dt < 16; ++dt) {
      const bf16* vp = &Vlds[cur][dt * 16 + l16][half * 16];
      const v8bf lo = *(const v8bf*)vp;
      const v8bf hi = *(const v8bf*)(vp + 8);
      v16bf vb;
#pragma unroll
      for (int i = 0; i < 8; ++i) { vb[i] = lo[i]; vb[i + 8] = hi[i]; }
      O[dt] = WMMA_BF16(pf, vb, O[dt]);
    }

    // drain our async copies into the next buffer, then block-wide handoff
    wait_async_done();
    __syncthreads();
  }

  // ---- epilogue: out = O / l ----
  float* orow = out + ((size_t)b * kT + q0) * kD;
#pragma unroll
  for (int r = 0; r < 8; ++r) {
    const float inv = 1.0f / l_run[r];
    const int row = r + 8 * half;
#pragma unroll
    for (int dt = 0; dt < 16; ++dt)
      orow[(size_t)row * kD + dt * 16 + l16] = O[dt][r] * inv;
  }
}

// ---------------------------------------------------------------------------
extern "C" void kernel_launch(void* const* d_in, const int* in_sizes, int n_in,
                              void* d_out, int out_size, void* d_ws, size_t ws_size,
                              hipStream_t stream) {
  (void)in_sizes; (void)n_in; (void)out_size; (void)ws_size;
  const float* x    = (const float*)d_in[0];   // [4,4096,256]
  const float* W    = (const float*)d_in[1];   // [768,256]
  const float* bias = (const float*)d_in[2];   // [768]
  const int*   mask = (const int*)d_in[3];     // [4,4096]
  float* out = (float*)d_out;

  const size_t elems = (size_t)kB * kT * kD;   // 4,194,304
  bf16* Qw = (bf16*)d_ws;
  bf16* Kw = Qw + elems;
  bf16* Vw = Kw + elems;                       // total 3*elems*2 = ~25 MB

  // QKV projection: one wave per 16x16 tile of the [B*T, 768] output.
  dim3 g1(kB * kT / 16, (3 * kD) / 16);        // (1024, 48)
  qkv_proj_kernel<<<g1, 32, 0, stream>>>(x, W, bias, Qw, Kw, Vw);

  // Flash attention: 256 blocks x 4 waves, each wave = 16 query rows.
  dim3 g2(kB * (kT / (kTQ * kNW)));            // 256
  attn_kernel<<<g2, kNW * 32, 0, stream>>>(Qw, Kw, Vw, mask, out);
}
